// GATQN_Bipartite_79817672229314
// MI455X (gfx1250) — compile-verified, matrix operand order checked by reading, not run on hardware
//
#include <hip/hip_runtime.h>

// ---------------- problem constants ----------------
#define B_   16
#define W_   128
#define T_   256
#define D_   128
#define WA_  64
#define TA_  128
#define EA_  32
#define EPS_ 1e-8f
#define SLOPE_ 0.2f

typedef __attribute__((ext_vector_type(16))) __bf16 v16bf;
typedef __attribute__((ext_vector_type(8)))  float  v8f;

union FragBF { v16bf v; __bf16 b[16]; unsigned short u[16]; uint4 q[2]; };
union FragF  { v8f v; float f[8]; };

__device__ inline float lrelu(float x) { return x > 0.f ? x : SLOPE_ * x; }

// A/B fragment K mapping for V_WMMA_*_16X16X32_BF16 (CDNA5 ISA 7.12.2):
// lane half h (lane>>4): elements j=0..7 -> K = h*8+j ; j=8..15 -> K = 16+h*8+(j-8)
__device__ inline int kmap(int h, int j) {
  return (j < 8) ? (h * 8 + j) : (16 + h * 8 + (j - 8));
}
__device__ inline void zero8(FragF& a) {
#pragma unroll
  for (int i = 0; i < 8; ++i) a.f[i] = 0.f;
}
__device__ inline void load_frag_q(const __bf16* p, FragBF& a) {
  const uint4* q = (const uint4*)p;
  a.q[0] = q[0]; a.q[1] = q[1];
}
// Load one 16x32 A-fragment row (this lane's row = lane&15) of edge_attribute (EA=32 floats).
__device__ inline void load_ea_frag(const float* ea, int hh, FragBF& a) {
#pragma unroll
  for (int j = 0; j < 8; ++j) a.b[j] = (__bf16)ea[hh * 8 + j];
#pragma unroll
  for (int j = 0; j < 8; ++j) a.b[8 + j] = (__bf16)ea[16 + hh * 8 + j];
}
__device__ inline v8f wmma_bf16(const FragBF& a, const FragBF& b, const FragF& c) {
  return __builtin_amdgcn_wmma_f32_16x16x32_bf16(false, a.v, false, b.v,
                                                 (short)0, c.v, false, false);
}

// ---------------- K0: pre-swizzle bf16 B-fragments (We, A2w/A2t/A2e) ----------------
// Fragment storage: frag[(kb*8+nt)][lane][16] bf16, contiguous 32B per lane.
__global__ void k_prep_frags(const float* __restrict__ Aw, const float* __restrict__ At,
                             const float* __restrict__ Ae, const float* __restrict__ We,
                             __bf16* __restrict__ fWe, __bf16* __restrict__ fAw,
                             __bf16* __restrict__ fAt, __bf16* __restrict__ fAe) {
  int gid = blockIdx.x * blockDim.x + threadIdx.x;
  if (gid >= 104 * 512) return;
  int frag = gid >> 9;          // 0..103
  int e    = gid & 511;         // lane*16 + j
  int lane = e >> 4, j = e & 15;
  int nl = lane & 15, h = lane >> 4;
  int kl = kmap(h, j);
  const float* src; __bf16* dst; int kb, nt, rowoff;
  if (frag < 8) {               // We: 32x128, kb=0
    src = We; dst = fWe + frag * 512; kb = 0; nt = frag; rowoff = 0;
  } else {
    int f2 = frag - 8;          // 0..95
    int mat = f2 / 32;
    int fi  = f2 % 32;          // kb*8 + nt
    kb = fi >> 3; nt = fi & 7;
    src = (mat == 0) ? Aw : ((mat == 1) ? At : Ae);
    dst = ((mat == 0) ? fAw : ((mat == 1) ? fAt : fAe)) + fi * 512;
    rowoff = 128;               // edge_vec block of the 384-row attention matrix
  }
  int K = kb * 32 + kl;
  int N = nt * 16 + nl;
  dst[e] = (__bf16)src[(size_t)(rowoff + K) * 128 + N];
}

// ---------------- K1: worker embeddings + wp_{w,t,e} ----------------
__global__ void k_worker_embed(const float* __restrict__ wattr, const float* __restrict__ Ww,
                               const float* __restrict__ bw_bias,
                               const float* __restrict__ Aw, const float* __restrict__ At,
                               const float* __restrict__ Ae,
                               float* __restrict__ worker_vec,
                               float* __restrict__ wpw, float* __restrict__ wpt,
                               float* __restrict__ wpe) {
  __shared__ float wa_s[WA_];
  __shared__ float wv_s[D_];
  int row = blockIdx.x;               // b*W + w
  int d = threadIdx.x;                // 0..127
  if (d < WA_) wa_s[d] = wattr[(size_t)row * WA_ + d];
  __syncthreads();
  float s = bw_bias[d];
  for (int k = 0; k < WA_; ++k) s += wa_s[k] * Ww[k * D_ + d];
  float wv = lrelu(s);
  worker_vec[(size_t)row * D_ + d] = wv;
  wv_s[d] = wv;
  __syncthreads();
  float sw = 0.f, st = 0.f, se = 0.f;
  for (int k = 0; k < D_; ++k) {
    float a = wv_s[k];
    sw += a * Aw[k * D_ + d];
    st += a * At[k * D_ + d];
    se += a * Ae[k * D_ + d];
  }
  wpw[(size_t)row * D_ + d] = sw;
  wpt[(size_t)row * D_ + d] = st;
  wpe[(size_t)row * D_ + d] = se;
}

// ---------------- K2: task embeddings + tp_{w,t,e} (raw attrs feed tp!) ----------------
__global__ void k_task_embed(const float* __restrict__ tattr, const float* __restrict__ Wt,
                             const float* __restrict__ bt_bias,
                             const float* __restrict__ Aw, const float* __restrict__ At,
                             const float* __restrict__ Ae,
                             float* __restrict__ task_vec,
                             float* __restrict__ tpw, float* __restrict__ tpt,
                             float* __restrict__ tpe) {
  __shared__ float ta_s[TA_];
  int row = blockIdx.x;               // b*T + t
  int d = threadIdx.x;
  ta_s[d] = tattr[(size_t)row * TA_ + d];
  __syncthreads();
  float s = bt_bias[d], sw = 0.f, st = 0.f, se = 0.f;
  for (int k = 0; k < TA_; ++k) {
    float a = ta_s[k];
    s  += a * Wt[k * D_ + d];
    sw += a * Aw[(256 + k) * D_ + d];
    st += a * At[(256 + k) * D_ + d];
    se += a * Ae[(256 + k) * D_ + d];
  }
  task_vec[(size_t)row * D_ + d] = lrelu(s);
  tpw[(size_t)row * D_ + d] = sw;
  tpt[(size_t)row * D_ + d] = st;
  tpe[(size_t)row * D_ + d] = se;
}

// ---------------- K3: edge scores (the hot WMMA kernel) ----------------
// One block per (b,w); 4 waves, each wave owns 16x(t-chunk) tiles.
// Low VGPR pressure: single 8-VGPR accumulator per N-tile, B-fragments
// streamed from L0/L2, outer loops kept rolled, allocation capped at 4 waves/EU.
__global__ __launch_bounds__(128)
__attribute__((amdgpu_waves_per_eu(4)))
void k_edge_scores(
    const float* __restrict__ eattr, const float* __restrict__ adj,
    const float* __restrict__ be_bias,
    const float* __restrict__ mw, const float* __restrict__ mt, const float* __restrict__ me,
    const float* __restrict__ wpw, const float* __restrict__ wpt, const float* __restrict__ wpe,
    const float* __restrict__ tpw, const float* __restrict__ tpt, const float* __restrict__ tpe,
    const __bf16* __restrict__ fWe,
    const __bf16* __restrict__ fAw, const __bf16* __restrict__ fAt,
    const __bf16* __restrict__ fAe,
    float* __restrict__ sw_out, float* __restrict__ st_out, float* __restrict__ se_out) {
  __shared__ __bf16 ev_s[4][16 * D_];           // per-wave edge_vec tile (bf16)
  __shared__ float h_s[4][16 * D_];             // per-wave pre-tanh tile (f32)
  __shared__ float wp_s[3][D_];
  __shared__ float be_s[D_];
  __shared__ float m_s[3][D_];
  int bw = blockIdx.x;                          // b*W + w
  int b  = bw >> 7;
  int tid = threadIdx.x;
  int wvid = tid >> 5, lane = tid & 31;
  int nl = lane & 15, hh = lane >> 4;

  wp_s[0][tid] = wpw[(size_t)bw * D_ + tid];
  wp_s[1][tid] = wpt[(size_t)bw * D_ + tid];
  wp_s[2][tid] = wpe[(size_t)bw * D_ + tid];
  be_s[tid] = be_bias[tid];
  m_s[0][tid] = mw[tid]; m_s[1][tid] = mt[tid]; m_s[2][tid] = me[tid];
  __syncthreads();

  __bf16* evp = ev_s[wvid];
  float* hp = h_s[wvid];

#pragma clang loop unroll(disable)
  for (int c = wvid; c < 16; c += 4) {
    int t0 = c * 16;
    const float* ea = eattr + ((size_t)bw * T_ + t0 + nl) * EA_;
    FragBF aE; load_ea_frag(ea, hh, aE);

    // GEMM1: edge_vec tile = lrelu(ea @ We + be) -> LDS (bf16, [m][k] layout)
#pragma unroll
    for (int nt = 0; nt < 8; ++nt) {
      FragBF bb; load_frag_q(fWe + nt * 512 + lane * 16, bb);
      FragF acc; zero8(acc);
      acc.v = wmma_bf16(aE, bb, acc);
      int N = nt * 16 + nl;
      float bias = be_s[N];
#pragma unroll
      for (int r = 0; r < 8; ++r)
        evp[(r + 8 * hh) * D_ + N] = (__bf16)lrelu(acc.f[r] + bias);
    }
    // Re-stage edge_vec tile as four K=32 A-fragments (K pairs are dword-aligned)
    FragBF aV[4];
    const unsigned int* ev32 = (const unsigned int*)evp;
#pragma unroll
    for (int kb = 0; kb < 4; ++kb) {
#pragma unroll
      for (int p = 0; p < 8; ++p) {
        int kl = kmap(hh, 2 * p);
        unsigned int w2 = ev32[(nl * D_ + kb * 32 + kl) >> 1];
        aV[kb].u[2 * p]     = (unsigned short)(w2 & 0xffffu);
        aV[kb].u[2 * p + 1] = (unsigned short)(w2 >> 16);
      }
    }
    // GEMM2 per branch: h = ev @ A2x ; then s = tanh(h + wp + tp) . m ; score = adj*exp(s)
#pragma clang loop unroll(disable)
    for (int x = 0; x < 3; ++x) {
      const __bf16* fA = (x == 0) ? fAw : ((x == 1) ? fAt : fAe);
      const float* tpx = (x == 0) ? tpw : ((x == 1) ? tpt : tpe);
#pragma unroll
      for (int nt = 0; nt < 8; ++nt) {
        FragF acc; zero8(acc);
#pragma unroll
        for (int kb = 0; kb < 4; ++kb) {
          FragBF bb; load_frag_q(fA + (kb * 8 + nt) * 512 + lane * 16, bb);
          acc.v = wmma_bf16(aV[kb], bb, acc);
        }
        int N = nt * 16 + nl;
#pragma unroll
        for (int r = 0; r < 8; ++r) hp[(r + 8 * hh) * D_ + N] = acc.f[r];
      }
      // Epilogue: all 32 lanes. Lane handles row m = lane&15, n-half = (lane>>4)*64.
      {
        int m = nl;
        const float* tprow = tpx + ((size_t)b * T_ + t0 + m) * D_;
        const float* wpx = wp_s[x];
        const float* mx  = m_s[x];
        int n0 = hh * 64;
        float s = 0.f;
#pragma clang loop unroll_count(8)
        for (int n = n0; n < n0 + 64; ++n) {
          float v = hp[m * D_ + n] + wpx[n] + tprow[n];
          s += tanhf(v) * mx[n];
        }
        s += __shfl_xor(s, 16, 32);
        if (lane < 16) {
          size_t row = (size_t)bw * T_ + t0 + m;
          float sc = adj[row] * __expf(s);
          float* so = (x == 0) ? sw_out : ((x == 1) ? st_out : se_out);
          so[row] = sc;
        }
      }
    }
  }
}

// ---------------- K4: normalization sums ----------------
__global__ void k_reduce_w(const float* __restrict__ sw, const float* __restrict__ se,
                           float* __restrict__ rw, float* __restrict__ re) {
  __shared__ float r1[T_], r2[T_];
  int bw = blockIdx.x; int t = threadIdx.x;
  size_t base = (size_t)bw * T_;
  r1[t] = sw[base + t]; r2[t] = se[base + t];
  __syncthreads();
  for (int s = T_ / 2; s > 0; s >>= 1) {
    if (t < s) { r1[t] += r1[t + s]; r2[t] += r2[t + s]; }
    __syncthreads();
  }
  if (t == 0) { rw[bw] = r1[0]; re[bw] = r2[0]; }
}
__global__ void k_reduce_t(const float* __restrict__ st, const float* __restrict__ se,
                           float* __restrict__ rt, float* __restrict__ re) {
  __shared__ float r1[W_], r2[W_];
  int bt = blockIdx.x; int b = bt >> 8; int t = bt & 255;
  int w = threadIdx.x;
  size_t idx = ((size_t)(b * W_ + w)) * T_ + t;
  r1[w] = st[idx]; r2[w] = se[idx];
  __syncthreads();
  for (int s = W_ / 2; s > 0; s >>= 1) {
    if (w < s) { r1[w] += r1[w + s]; r2[w] += r2[w + s]; }
    __syncthreads();
  }
  if (w == 0) { rt[bt] = r1[0]; re[bt] = r2[0]; }
}

// ---------------- K5: worker aggregation + head ----------------
__global__ __launch_bounds__(128)
__attribute__((amdgpu_waves_per_eu(4)))
void k_worker_out(
    const float* __restrict__ eattr, const float* __restrict__ be_bias,
    const float* __restrict__ sw, const float* __restrict__ se,
    const float* __restrict__ rw, const float* __restrict__ rew,
    const float* __restrict__ task_vec, const float* __restrict__ worker_vec,
    const __bf16* __restrict__ fWe,
    const float* __restrict__ Gw, const float* __restrict__ gbw,
    float* __restrict__ out) {
  __shared__ float alpha_w[T_], alpha_e[T_];
  __shared__ float wvec[D_];
  __shared__ float agg[3 * D_];
  int bw = blockIdx.x; int b = bw >> 7;
  int tid = threadIdx.x, wvid = tid >> 5, lane = tid & 31;
  int nl = lane & 15, hh = lane >> 4;
  float invw = 1.f / fmaxf(rw[bw], EPS_);
  float inve = 1.f / fmaxf(rew[bw], EPS_);
  size_t base = (size_t)bw * T_;
  for (int t = tid; t < T_; t += 128) {
    alpha_w[t] = sw[base + t] * invw;
    alpha_e[t] = se[base + t] * inve;
  }
  wvec[tid] = 0.f;
  __syncthreads();

  float part[8];
#pragma unroll
  for (int i = 0; i < 8; ++i) part[i] = 0.f;

#pragma clang loop unroll(disable)
  for (int c = wvid; c < 16; c += 4) {
    int t0 = c * 16;
    const float* ea = eattr + (base + t0 + nl) * EA_;
    FragBF aE; load_ea_frag(ea, hh, aE);
#pragma unroll
    for (int nt = 0; nt < 8; ++nt) {
      FragBF bb; load_frag_q(fWe + nt * 512 + lane * 16, bb);
      FragF acc; zero8(acc);
      acc.v = wmma_bf16(aE, bb, acc);
      int N = nt * 16 + nl;
      float bias = be_bias[N];
      float p = 0.f;
#pragma unroll
      for (int r = 0; r < 8; ++r)
        p += lrelu(acc.f[r] + bias) * alpha_e[t0 + r + 8 * hh];
      part[nt] += p;
    }
  }
#pragma unroll
  for (int nt = 0; nt < 8; ++nt) {
    float p = part[nt] + __shfl_xor(part[nt], 16, 32);
    if (lane < 16) atomicAdd(&wvec[nt * 16 + lane], p);
  }
  __syncthreads();

  int d = tid;
  float tmpw = 0.f;
#pragma clang loop unroll_count(8)
  for (int t = 0; t < T_; ++t)
    tmpw += alpha_w[t] * task_vec[((size_t)b * T_ + t) * D_ + d];
  agg[d]          = worker_vec[(size_t)bw * D_ + d];
  agg[D_ + d]     = tmpw;
  agg[2 * D_ + d] = wvec[d];
  __syncthreads();
  float o = gbw[d];
#pragma clang loop unroll_count(8)
  for (int k = 0; k < 3 * D_; ++k) o += agg[k] * Gw[k * D_ + d];
  out[(size_t)bw * D_ + d] = lrelu(o);
}

// ---------------- K6: task aggregation + head ----------------
__global__ __launch_bounds__(128)
__attribute__((amdgpu_waves_per_eu(4)))
void k_task_out(
    const float* __restrict__ eattr, const float* __restrict__ be_bias,
    const float* __restrict__ st, const float* __restrict__ se,
    const float* __restrict__ rt, const float* __restrict__ ret,
    const float* __restrict__ worker_vec, const float* __restrict__ task_vec,
    const __bf16* __restrict__ fWe,
    const float* __restrict__ Gt, const float* __restrict__ gbt,
    float* __restrict__ out) {
  __shared__ float alpha_t[W_], alpha_e[W_];
  __shared__ float wvec[D_];
  __shared__ float agg[3 * D_];
  int bt = blockIdx.x; int b = bt >> 8; int t = bt & 255;
  int tid = threadIdx.x, wvid = tid >> 5, lane = tid & 31;
  int nl = lane & 15, hh = lane >> 4;
  float invt = 1.f / fmaxf(rt[bt], EPS_);
  float inve = 1.f / fmaxf(ret[bt], EPS_);
  if (tid < W_) {
    size_t idx = ((size_t)(b * W_ + tid)) * T_ + t;
    alpha_t[tid] = st[idx] * invt;
    alpha_e[tid] = se[idx] * inve;
  }
  wvec[tid] = 0.f;
  __syncthreads();

  float part[8];
#pragma unroll
  for (int i = 0; i < 8; ++i) part[i] = 0.f;

#pragma clang loop unroll(disable)
  for (int c = wvid; c < 8; c += 4) {   // 8 chunks of 16 workers, 2 per wave
    int w0 = c * 16;
    const float* ea = eattr + (((size_t)(b * W_ + w0 + nl)) * T_ + t) * EA_;
    FragBF aE; load_ea_frag(ea, hh, aE);
#pragma unroll
    for (int nt = 0; nt < 8; ++nt) {
      FragBF bb; load_frag_q(fWe + nt * 512 + lane * 16, bb);
      FragF acc; zero8(acc);
      acc.v = wmma_bf16(aE, bb, acc);
      int N = nt * 16 + nl;
      float bias = be_bias[N];
      float p = 0.f;
#pragma unroll
      for (int r = 0; r < 8; ++r)
        p += lrelu(acc.f[r] + bias) * alpha_e[w0 + r + 8 * hh];
      part[nt] += p;
    }
  }
#pragma unroll
  for (int nt = 0; nt < 8; ++nt) {
    float p = part[nt] + __shfl_xor(part[nt], 16, 32);
    if (lane < 16) atomicAdd(&wvec[nt * 16 + lane], p);
  }
  __syncthreads();

  int d = tid;
  float tmpt = 0.f;
#pragma clang loop unroll_count(8)
  for (int w = 0; w < W_; ++w)
    tmpt += alpha_t[w] * worker_vec[((size_t)(b * W_ + w)) * D_ + d];
  agg[d]          = task_vec[(size_t)bt * D_ + d];
  agg[D_ + d]     = tmpt;
  agg[2 * D_ + d] = wvec[d];
  __syncthreads();
  float o = gbt[d];
#pragma clang loop unroll_count(8)
  for (int k = 0; k < 3 * D_; ++k) o += agg[k] * Gt[k * D_ + d];
  out[(size_t)bt * D_ + d] = lrelu(o);
}

// ---------------- launcher ----------------
extern "C" void kernel_launch(void* const* d_in, const int* in_sizes, int n_in,
                              void* d_out, int out_size, void* d_ws, size_t ws_size,
                              hipStream_t stream) {
  const float* adj   = (const float*)d_in[0];
  const float* wattr = (const float*)d_in[1];
  const float* tattr = (const float*)d_in[2];
  const float* eattr = (const float*)d_in[3];
  const float* Ww = (const float*)d_in[4];  const float* bwv = (const float*)d_in[5];
  const float* Wt = (const float*)d_in[6];  const float* btv = (const float*)d_in[7];
  const float* We = (const float*)d_in[8];  const float* bev = (const float*)d_in[9];
  const float* Aw = (const float*)d_in[10]; const float* At  = (const float*)d_in[11];
  const float* Ae = (const float*)d_in[12];
  const float* mw = (const float*)d_in[13]; const float* mt  = (const float*)d_in[14];
  const float* me = (const float*)d_in[15];
  const float* Gw = (const float*)d_in[16]; const float* gbw = (const float*)d_in[17];
  const float* Gt = (const float*)d_in[18]; const float* gbt = (const float*)d_in[19];
  float* out = (float*)d_out;

  char* p = (char*)d_ws;
  auto carve = [&](size_t bytes) -> void* {
    void* r = (void*)p;
    p += (bytes + 255) & ~(size_t)255;
    return r;
  };
  __bf16* fWe = (__bf16*)carve(8 * 512 * 2);
  __bf16* fAw = (__bf16*)carve(32 * 512 * 2);
  __bf16* fAt = (__bf16*)carve(32 * 512 * 2);
  __bf16* fAe = (__bf16*)carve(32 * 512 * 2);
  float* worker_vec = (float*)carve((size_t)B_ * W_ * D_ * 4);
  float* task_vec   = (float*)carve((size_t)B_ * T_ * D_ * 4);
  float* wpw = (float*)carve((size_t)B_ * W_ * D_ * 4);
  float* wpt = (float*)carve((size_t)B_ * W_ * D_ * 4);
  float* wpe = (float*)carve((size_t)B_ * W_ * D_ * 4);
  float* tpw = (float*)carve((size_t)B_ * T_ * D_ * 4);
  float* tpt = (float*)carve((size_t)B_ * T_ * D_ * 4);
  float* tpe = (float*)carve((size_t)B_ * T_ * D_ * 4);
  float* s_w = (float*)carve((size_t)B_ * W_ * T_ * 4);
  float* s_t = (float*)carve((size_t)B_ * W_ * T_ * 4);
  float* s_e = (float*)carve((size_t)B_ * W_ * T_ * 4);
  float* rw  = (float*)carve((size_t)B_ * W_ * 4);
  float* rew = (float*)carve((size_t)B_ * W_ * 4);
  float* rt  = (float*)carve((size_t)B_ * T_ * 4);
  float* ret = (float*)carve((size_t)B_ * T_ * 4);

  k_prep_frags<<<(104 * 512 + 255) / 256, 256, 0, stream>>>(Aw, At, Ae, We,
                                                            fWe, fAw, fAt, fAe);
  k_worker_embed<<<B_ * W_, 128, 0, stream>>>(wattr, Ww, bwv, Aw, At, Ae,
                                              worker_vec, wpw, wpt, wpe);
  k_task_embed<<<B_ * T_, 128, 0, stream>>>(tattr, Wt, btv, Aw, At, Ae,
                                            task_vec, tpw, tpt, tpe);
  k_edge_scores<<<B_ * W_, 128, 0, stream>>>(eattr, adj, bev, mw, mt, me,
                                             wpw, wpt, wpe, tpw, tpt, tpe,
                                             fWe, fAw, fAt, fAe,
                                             s_w, s_t, s_e);
  k_reduce_w<<<B_ * W_, T_, 0, stream>>>(s_w, s_e, rw, rew);
  k_reduce_t<<<B_ * T_, W_, 0, stream>>>(s_t, s_e, rt, ret);
  k_worker_out<<<B_ * W_, 128, 0, stream>>>(eattr, bev, s_w, s_e, rw, rew,
                                            task_vec, worker_vec, fWe, Gw, gbw,
                                            out);
  k_task_out<<<B_ * T_, 128, 0, stream>>>(eattr, bev, s_t, s_e, rt, ret,
                                          worker_vec, task_vec, fWe, Gt, gbt,
                                          out + (size_t)B_ * W_ * D_);
}
